// TransInrgan_22445499088948
// MI455X (gfx1250) — matrix operand order repaired and checked by reading, not compiled, not run on hardware
//
#include <hip/hip_runtime.h>
#include <hip/hip_bf16.h>
#include <cstdint>

typedef _Float16 v16h __attribute__((ext_vector_type(16)));
typedef float    v8f  __attribute__((ext_vector_type(8)));

#define DEV static __device__ __forceinline__

DEV v8f wmma_f16(v16h a, v16h b, v8f c) {
  return __builtin_amdgcn_wmma_f32_16x16x32_f16(false, a, false, b, (short)0, c, false, false);
}

DEV v16h frag_from2(uint4 u0, uint4 u1) {
  union { uint4 u[2]; v16h v; } c;
  c.u[0] = u0; c.u[1] = u1;
  return c.v;
}

// A-fragment (16x32 f16): two contiguous 16B runs per lane.
DEV v16h load_frag_a(const _Float16* p /* row ptr at k0 */, int hf) {
  const uint4 u0 = *(const uint4*)(p + hf * 16);
  const uint4 u1 = *(const uint4*)(p + 8 + hf * 16);
  return frag_from2(u0, u1);
}

// B-fragment (32x16 f16) from K-contiguous storage: one 32B run per lane.
DEV v16h load_frag_b(const _Float16* p /* col ptr at k0 + hf*16 */) {
  const uint4* q = (const uint4*)p;
  return frag_from2(q[0], q[1]);
}

// ---------------------------------------------------------------------------
// fp32 [K,N] -> f16 transposed [N,K], tiled through LDS for coalescing.
// ---------------------------------------------------------------------------
__global__ __launch_bounds__(256) void cvt_t_kernel(const float* __restrict__ W,
                                                    _Float16* __restrict__ Wt,
                                                    int K, int N) {
  __shared__ float tile[32][33];
  int k0 = blockIdx.y * 32, n0 = blockIdx.x * 32;
  int tx = threadIdx.x, ty = threadIdx.y;  // 32 x 8
#pragma unroll
  for (int j = 0; j < 32; j += 8) {
    int k = k0 + ty + j, n = n0 + tx;
    tile[ty + j][tx] = (k < K && n < N) ? W[(size_t)k * N + n] : 0.0f;
  }
  __syncthreads();
#pragma unroll
  for (int j = 0; j < 32; j += 8) {
    int n = n0 + ty + j, k = k0 + tx;
    if (n < N && k < K) Wt[(size_t)n * K + k] = (_Float16)tile[tx][ty + j];
  }
}

// ---------------------------------------------------------------------------
// Patchify: data [32,3,128,128] -> patches f16 [8224,192]; wtoken rows zeroed
// ---------------------------------------------------------------------------
__global__ void patchify_kernel(const float* __restrict__ data, _Float16* __restrict__ p) {
  long i = (long)blockIdx.x * blockDim.x + threadIdx.x;
  if (i >= 8224L * 192) return;
  int e = (int)(i % 192);
  long g = i / 192;
  int b = (int)(g / 257), t = (int)(g % 257);
  _Float16 val = (_Float16)0.0f;
  if (t < 256) {
    int ph = t >> 4, pw = t & 15;
    int ch = e % 3, pp = e / 3;
    int py = pp >> 3, px = pp & 7;
    val = (_Float16)data[(((size_t)b * 3 + ch) * 128 + (size_t)ph * 8 + py) * 128 + pw * 8 + px];
  }
  p[i] = val;
}

__global__ void set_wtokens_kernel(const float* __restrict__ wt, float* __restrict__ x) {
  int i = blockIdx.x * blockDim.x + threadIdx.x;
  if (i >= 32 * 512) return;
  int b = i >> 9, c = i & 511;
  x[((size_t)b * 257 + 256) * 512 + c] = wt[c];
}

// ---------------------------------------------------------------------------
// V-transpose: qkv16 V-section -> Vt [32][8][64][288] f16 (K zero-padded)
// ---------------------------------------------------------------------------
__global__ void vt_kernel(const _Float16* __restrict__ qkv, _Float16* __restrict__ Vt) {
  long i = (long)blockIdx.x * blockDim.x + threadIdx.x;
  if (i >= 32L * 8 * 64 * 288) return;
  int k = (int)(i % 288);
  long r = i / 288;
  int n = (int)(r & 63);
  long bh = r >> 6;
  int h = (int)(bh & 7);
  int b = (int)(bh >> 3);
  _Float16 v = (_Float16)0.0f;
  if (k < 257) v = qkv[((size_t)b * 257 + k) * 1536 + 1024 + h * 64 + n];
  Vt[i] = v;
}

// ---------------------------------------------------------------------------
// WMMA GEMM with transposed B (Bt[N][K], K-contiguous).
// 256 threads = 8 waves; block tile 256x64; wave tile 32x64 (8 WMMAs/K-step).
// Double-buffered LDS staging, one barrier per K-step, branch-free loads
// (indices clamped; garbage lanes masked in the epilogue).
// ---------------------------------------------------------------------------
__global__ __launch_bounds__(256) void gemm_f16_kernel(
    const _Float16* __restrict__ A, const _Float16* __restrict__ Bt,
    const float* __restrict__ bias, const float* __restrict__ res,
    float* __restrict__ out32, _Float16* __restrict__ out16,
    int M, int N, int K, int do_gelu) {
  __shared__ _Float16 Bs[2][64 * 40];
  const int lane = threadIdx.x & 31;
  const int wave = threadIdx.x >> 5;
  const int hf = lane >> 4;
  const int nn = lane & 15;
  const int n0 = blockIdx.x * 64;
  const int m0 = blockIdx.y * 256 + wave * 32;

  // M is always a multiple of 32 here -> row validity is wave-uniform.
  const bool row0v = m0 < M;
  const bool row1v = m0 + 16 < M;
  // Clamp rows: invalid waves read row 0; their stores are masked below.
  const int ar0 = row0v ? (m0 + nn) : 0;
  const int ar1 = row1v ? (m0 + 16 + nn) : 0;
  const _Float16* a0p = A + (size_t)ar0 * K;
  const _Float16* a1p = A + (size_t)ar1 * K;

  // Staging: 256 threads cover 64 cols x 32 K halves, one uint4 (8 halves) each.
  const int sn = threadIdx.x >> 2;           // 0..63 : B column within tile
  const int seg = (threadIdx.x & 3) * 8;     // 0/8/16/24 halves
  // Clamp column: OOB columns stage garbage that only reaches masked outputs.
  const int snc = (n0 + sn < N) ? (n0 + sn) : (N - 1);
  const _Float16* bsrc = Bt + (size_t)snc * K + seg;

  v8f acc[2][4];
#pragma unroll
  for (int r = 0; r < 2; ++r)
#pragma unroll
    for (int t = 0; t < 4; ++t)
#pragma unroll
      for (int i = 0; i < 8; ++i) acc[r][t][i] = 0.0f;

  // Prologue: stage k0 = 0 into buffer 0.
  *(uint4*)(&Bs[0][sn * 40 + seg]) = *(const uint4*)bsrc;

  int p = 0;
  for (int k0 = 0; k0 < K; k0 += 32) {
    __syncthreads();
    const bool more = (k0 + 32 < K);  // wave-uniform scalar branch
    // Next B tile into registers (overlaps with WMMA below).
    uint4 nu = make_uint4(0, 0, 0, 0);
    if (more) nu = *(const uint4*)(bsrc + k0 + 32);

    // Stream-prefetch A ahead (global_prefetch_b8).
    __builtin_prefetch((const void*)(a0p + k0 + 64), 0, 0);

    // A fragments: unconditional (clamped pointers).
    v16h a0 = load_frag_a(a0p + k0, hf);
    v16h a1 = load_frag_a(a1p + k0, hf);

    // All 4 B fragments, then an 8-WMMA burst.
    v16h bfr[4];
#pragma unroll
    for (int t = 0; t < 4; ++t) bfr[t] = load_frag_b(&Bs[p][(t * 16 + nn) * 40 + hf * 16]);
#pragma unroll
    for (int t = 0; t < 4; ++t) {
      acc[0][t] = wmma_f16(a0, bfr[t], acc[0][t]);
      acc[1][t] = wmma_f16(a1, bfr[t], acc[1][t]);
    }

    if (more) *(uint4*)(&Bs[p ^ 1][sn * 40 + seg]) = nu;
    p ^= 1;
  }

  // Epilogue: bias -> gelu -> +res -> store (f32 and/or f16)
#pragma unroll
  for (int r = 0; r < 2; ++r) {
    if (r == 0 ? !row0v : !row1v) continue;
#pragma unroll
    for (int t = 0; t < 4; ++t) {
      int col = n0 + t * 16 + nn;
      if (col >= N) continue;
#pragma unroll
      for (int i = 0; i < 8; ++i) {
        int row = m0 + r * 16 + i + hf * 8;
        float vv = acc[r][t][i];
        if (bias) vv += bias[col];
        if (do_gelu) {
          float x3 = vv * vv * vv;
          vv = 0.5f * vv * (1.0f + tanhf(0.7978845608028654f * (vv + 0.044715f * x3)));
        }
        size_t idx = (size_t)row * N + col;
        if (res) vv += res[idx];
        if (out32) out32[idx] = vv;
        if (out16) out16[idx] = (_Float16)vv;
      }
    }
  }
}

// ---------------------------------------------------------------------------
// LayerNorm (512 cols) -> f16 output. rowOff/rowStride select source rows.
// ---------------------------------------------------------------------------
__global__ __launch_bounds__(128) void ln_f16_kernel(
    const float* __restrict__ x, const float* __restrict__ g, const float* __restrict__ bb,
    _Float16* __restrict__ out, int rows, int rowOff, int rowStride) {
  int rr = blockIdx.x;
  if (rr >= rows) return;
  size_t row = (size_t)rowOff + (size_t)rr * rowStride;
  const float* xr = x + row * 512;
  float v[4];
  float s = 0.0f, s2 = 0.0f;
#pragma unroll
  for (int j = 0; j < 4; ++j) {
    v[j] = xr[threadIdx.x + j * 128];
    s += v[j];
    s2 += v[j] * v[j];
  }
#pragma unroll
  for (int o = 16; o > 0; o >>= 1) {
    s += __shfl_down(s, o, 32);
    s2 += __shfl_down(s2, o, 32);
  }
  __shared__ float red[8];
  if ((threadIdx.x & 31) == 0) {
    red[threadIdx.x >> 5] = s;
    red[4 + (threadIdx.x >> 5)] = s2;
  }
  __syncthreads();
  float ts = red[0] + red[1] + red[2] + red[3];
  float ts2 = red[4] + red[5] + red[6] + red[7];
  float mu = ts * (1.0f / 512.0f);
  float var = ts2 * (1.0f / 512.0f) - mu * mu;
  float inv = rsqrtf(var + 1e-5f);
#pragma unroll
  for (int j = 0; j < 4; ++j) {
    int c = threadIdx.x + j * 128;
    out[(size_t)rr * 512 + c] = (_Float16)((v[j] - mu) * inv * g[c] + bb[c]);
  }
}

// ---------------------------------------------------------------------------
// Fused attention per (16-row tile, head, batch): one wave.
// S = Q K^T in VGPRs (17 tiles x 8 f32), shfl softmax, P in LDS, O = P Vt.
// All fragments are 128-bit vector loads.
// ---------------------------------------------------------------------------
__global__ __launch_bounds__(32) void attn_f16_kernel(
    const _Float16* __restrict__ qkv, const _Float16* __restrict__ Vt,
    _Float16* __restrict__ o16) {
  const int m0 = blockIdx.x * 16;
  const int h = blockIdx.y;
  const int b = blockIdx.z;
  const int lane = threadIdx.x;
  const int hf = lane >> 4, nn = lane & 15;
  __shared__ _Float16 P[16 * 288];
  const size_t rowBase = (size_t)b * 257;
  const int qc = h * 64, kc = 512 + h * 64;

  int arow = m0 + nn;
  if (arow > 256) arow = 256;
  v16h aq[2];
#pragma unroll
  for (int kk = 0; kk < 2; ++kk)
    aq[kk] = load_frag_a(qkv + (rowBase + arow) * 1536 + qc + kk * 32, hf);

  float s[17][8];
#pragma unroll
  for (int t = 0; t < 17; ++t) {
    v8f acc;
#pragma unroll
    for (int i = 0; i < 8; ++i) acc[i] = 0.0f;
    int kr = t * 16 + nn;  // column of K^T == row of K
    if (kr > 256) kr = 256;
    v16h bf0 = load_frag_b(qkv + (rowBase + kr) * 1536 + kc + hf * 16);
    v16h bf1 = load_frag_b(qkv + (rowBase + kr) * 1536 + kc + 32 + hf * 16);
    acc = wmma_f16(aq[0], bf0, acc);
    acc = wmma_f16(aq[1], bf1, acc);
#pragma unroll
    for (int i = 0; i < 8; ++i) s[t][i] = acc[i];
  }

  // softmax over 257 columns (scale 1/sqrt(64)=0.125), masked past 256
#pragma unroll
  for (int i = 0; i < 8; ++i) {
    float mv = -3.0e38f;
#pragma unroll
    for (int t = 0; t < 17; ++t) {
      int col = t * 16 + nn;
      s[t][i] = (col < 257) ? s[t][i] * 0.125f : -3.0e38f;
      mv = fmaxf(mv, s[t][i]);
    }
#pragma unroll
    for (int o = 1; o < 16; o <<= 1) mv = fmaxf(mv, __shfl_xor(mv, o, 32));
    float sm = 0.0f;
#pragma unroll
    for (int t = 0; t < 17; ++t) {
      float e = __expf(s[t][i] - mv);
      s[t][i] = e;
      sm += e;
    }
#pragma unroll
    for (int o = 1; o < 16; o <<= 1) sm += __shfl_xor(sm, o, 32);
    float inv = 1.0f / sm;
    int prow = i + hf * 8;
#pragma unroll
    for (int t = 0; t < 17; ++t) P[prow * 288 + t * 16 + nn] = (_Float16)(s[t][i] * inv);
  }
  {  // zero K-pad columns 272..287
    int r = nn, c0 = 272 + hf * 8;
#pragma unroll
    for (int j = 0; j < 8; ++j) P[r * 288 + c0 + j] = (_Float16)0.0f;
  }
  __syncthreads();

  v8f oacc[4];
#pragma unroll
  for (int t = 0; t < 4; ++t)
#pragma unroll
    for (int i = 0; i < 8; ++i) oacc[t][i] = 0.0f;

  const _Float16* vtb = Vt + (size_t)(b * 8 + h) * 64 * 288;
  for (int k0 = 0; k0 < 288; k0 += 32) {
    v16h ap = load_frag_a(P + nn * 288 + k0, hf);
    v16h bv[4];
#pragma unroll
    for (int t = 0; t < 4; ++t)
      bv[t] = load_frag_b(vtb + (size_t)(t * 16 + nn) * 288 + k0 + hf * 16);
#pragma unroll
    for (int t = 0; t < 4; ++t) oacc[t] = wmma_f16(ap, bv[t], oacc[t]);
  }
#pragma unroll
  for (int t = 0; t < 4; ++t) {
#pragma unroll
    for (int i = 0; i < 8; ++i) {
      int row = m0 + i + hf * 8;
      if (row < 257) o16[(rowBase + row) * 512 + h * 64 + t * 16 + nn] = (_Float16)oacc[t][i];
    }
  }
}

// ---------------------------------------------------------------------------
// Hyponet modulation: low-rank hw1@hw2 -> sigmoid * base, concat bias rows.
// ---------------------------------------------------------------------------
__global__ void modulate_kernel(const float* __restrict__ mv,
                                const float* __restrict__ b0, const float* __restrict__ b1,
                                const float* __restrict__ b2, const float* __restrict__ b3,
                                const float* __restrict__ b4, float* __restrict__ out) {
  const int lls[5] = {0, 772, 6148, 11524, 16900};
  const int ns[5] = {2, 256, 256, 256, 256};
  const int ms[5] = {256, 256, 256, 256, 3};
  const int rs[5] = {2, 10, 10, 10, 3};
  const int offs[5] = {0, 768, 66560, 132352, 198144};
  long i = (long)blockIdx.x * blockDim.x + threadIdx.x;
  if (i >= 32L * 198915) return;
  int b = (int)(i / 198915);
  int pos = (int)(i - (long)b * 198915);
  int L = 4;
  if (pos < offs[1]) L = 0;
  else if (pos < offs[2]) L = 1;
  else if (pos < offs[3]) L = 2;
  else if (pos < offs[4]) L = 3;
  int local = pos - offs[L];
  int n = ns[L], m = ms[L], r = rs[L], ll = lls[L];
  int ii = local / m, j = local - ii * m;
  const float* mvb = mv + (size_t)b * 17680;
  float o;
  if (ii == n) {
    o = mvb[ll + n * r + r * m + j];
  } else {
    const float* h1 = mvb + ll + ii * r;
    const float* h2 = mvb + ll + n * r + j;
    float acc = 0.0f;
    for (int k = 0; k < r; ++k) acc += h1[k] * h2[(size_t)k * m];
    float sg = 1.0f / (1.0f + __expf(-acc));
    const float* base = (L == 0) ? b0 : (L == 1) ? b1 : (L == 2) ? b2 : (L == 3) ? b3 : b4;
    o = base[ii * m + j] * sg;
  }
  out[i] = o;
}

// ---------------------------------------------------------------------------
extern "C" void kernel_launch(void* const* d_in, const int* in_sizes, int n_in,
                              void* d_out, int out_size, void* d_ws, size_t ws_size,
                              hipStream_t stream) {
  (void)in_sizes; (void)n_in; (void)out_size; (void)ws_size;
  const float* data = (const float*)d_in[0];
  const float* tok_w = (const float*)d_in[1];
  const float* tok_b = (const float*)d_in[2];
  const float* wtokens = (const float*)d_in[3];
  const float* ln1_g = (const float*)d_in[4];
  const float* ln1_b = (const float*)d_in[5];
  const float* wqkv = (const float*)d_in[6];
  const float* bqkv = (const float*)d_in[7];
  const float* wo = (const float*)d_in[8];
  const float* bo = (const float*)d_in[9];
  const float* ln2_g = (const float*)d_in[10];
  const float* ln2_b = (const float*)d_in[11];
  const float* w1 = (const float*)d_in[12];
  const float* b1 = (const float*)d_in[13];
  const float* w2 = (const float*)d_in[14];
  const float* b2 = (const float*)d_in[15];
  const float* mod_ln_g = (const float*)d_in[16];
  const float* mod_ln_b = (const float*)d_in[17];
  const float* mod_w = (const float*)d_in[18];
  const float* mod_b = (const float*)d_in[19];
  const float* base0 = (const float*)d_in[20];
  const float* base1 = (const float*)d_in[21];
  const float* base2 = (const float*)d_in[22];
  const float* base3 = (const float*)d_in[23];
  const float* base4 = (const float*)d_in[24];
  float* out = (float*)d_out;

  // workspace carve-out
  char* ws = (char*)d_ws;
  size_t off = 0;
  auto alloc = [&](size_t bytes) -> void* {
    void* p = ws + off;
    off += (bytes + 255) & ~(size_t)255;
    return p;
  };
  const long R = 8224;  // 32 * 257 token rows
  _Float16* tokwT = (_Float16*)alloc(512L * 192 * 2);          // [512][192]
  _Float16* wqkvT = (_Float16*)alloc(6L * 1536 * 512 * 2);     // per layer [1536][512]
  _Float16* woT = (_Float16*)alloc(6L * 512 * 512 * 2);        // [512][512]
  _Float16* w1T = (_Float16*)alloc(6L * 2048 * 512 * 2);       // [2048][512]
  _Float16* w2T = (_Float16*)alloc(6L * 512 * 2048 * 2);       // [512][2048]
  _Float16* modwT = (_Float16*)alloc(17680L * 512 * 2);        // [17680][512]
  _Float16* patch16 = (_Float16*)alloc(R * 192 * 2);
  float* x32 = (float*)alloc(R * 512 * 4);
  _Float16* bufA16 = (_Float16*)alloc(R * 512 * 2);            // LN out / attention out
  _Float16* qkv16 = (_Float16*)alloc(R * 1536 * 2);
  _Float16* ff16 = (_Float16*)alloc(R * 2048 * 2);
  _Float16* Vt16 = (_Float16*)alloc(32L * 8 * 64 * 288 * 2);
  _Float16* lnlast16 = (_Float16*)alloc(32L * 512 * 2);
  float* modvec32 = (float*)alloc(32L * 17680 * 4);

  auto cvtT = [&](const float* src, _Float16* dst, int K, int N) {
    dim3 g((N + 31) / 32, (K + 31) / 32);
    cvt_t_kernel<<<g, dim3(32, 8), 0, stream>>>(src, dst, K, N);
  };
  cvtT(tok_w, tokwT, 192, 512);
  for (int l = 0; l < 6; ++l) {
    cvtT(wqkv + (size_t)l * 512 * 1536, wqkvT + (size_t)l * 1536 * 512, 512, 1536);
    cvtT(wo + (size_t)l * 512 * 512, woT + (size_t)l * 512 * 512, 512, 512);
    cvtT(w1 + (size_t)l * 512 * 2048, w1T + (size_t)l * 2048 * 512, 512, 2048);
    cvtT(w2 + (size_t)l * 2048 * 512, w2T + (size_t)l * 512 * 2048, 2048, 512);
  }
  cvtT(mod_w, modwT, 512, 17680);

  patchify_kernel<<<(int)((R * 192 + 255) / 256), 256, 0, stream>>>(data, patch16);

  const int GY = 33;  // ceil(8224/256)
  // tokenize: x = patches @ tok_w + tok_b
  gemm_f16_kernel<<<dim3(8, GY), 256, 0, stream>>>(patch16, tokwT, tok_b, nullptr,
                                                   x32, nullptr, 8224, 512, 192, 0);
  set_wtokens_kernel<<<64, 256, 0, stream>>>(wtokens, x32);

  for (int l = 0; l < 6; ++l) {
    ln_f16_kernel<<<8224, 128, 0, stream>>>(x32, ln1_g + l * 512, ln1_b + l * 512, bufA16,
                                            8224, 0, 1);
    gemm_f16_kernel<<<dim3(24, GY), 256, 0, stream>>>(bufA16, wqkvT + (size_t)l * 1536 * 512,
                                                      bqkv + l * 1536, nullptr, nullptr, qkv16,
                                                      8224, 1536, 512, 0);
    vt_kernel<<<(int)((32L * 8 * 64 * 288 + 255) / 256), 256, 0, stream>>>(qkv16, Vt16);
    attn_f16_kernel<<<dim3(17, 8, 32), 32, 0, stream>>>(qkv16, Vt16, bufA16);
    gemm_f16_kernel<<<dim3(8, GY), 256, 0, stream>>>(bufA16, woT + (size_t)l * 512 * 512,
                                                     bo + l * 512, x32, x32, nullptr,
                                                     8224, 512, 512, 0);
    ln_f16_kernel<<<8224, 128, 0, stream>>>(x32, ln2_g + l * 512, ln2_b + l * 512, bufA16,
                                            8224, 0, 1);
    gemm_f16_kernel<<<dim3(32, GY), 256, 0, stream>>>(bufA16, w1T + (size_t)l * 2048 * 512,
                                                      b1 + l * 2048, nullptr, nullptr, ff16,
                                                      8224, 2048, 512, 1);
    gemm_f16_kernel<<<dim3(8, GY), 256, 0, stream>>>(ff16, w2T + (size_t)l * 512 * 2048,
                                                     b2 + l * 512, x32, x32, nullptr,
                                                     8224, 512, 2048, 0);
  }

  // modvec = LN(x[:, -1, :]) @ mod_w + mod_b
  ln_f16_kernel<<<32, 128, 0, stream>>>(x32, mod_ln_g, mod_ln_b, lnlast16, 32, 256, 257);
  gemm_f16_kernel<<<dim3(277, 1), 256, 0, stream>>>(lnlast16, modwT, mod_b, nullptr,
                                                    modvec32, nullptr, 32, 17680, 512, 0);

  modulate_kernel<<<(int)((32L * 198915 + 255) / 256), 256, 0, stream>>>(
      modvec32, base0, base1, base2, base3, base4, out);
}
